// EmbeddingBank_11862699671789
// MI455X (gfx1250) — compile-verified, matrix operand order checked by reading, not run on hardware
//
#include <hip/hip_runtime.h>

// ---------------- problem constants (match reference) ----------------
#define BANK_N   1000000
#define EMB      128
#define NQ       256
#define TOPK     16
// ---------------- tiling ----------------
#define CHUNK    4096                           // bank rows per block
#define NBLK     ((BANK_N + CHUNK - 1) / CHUNK) // = 245
#define ROWS_IT  32                             // bank rows per pipeline stage
#define NIT      (CHUNK / ROWS_IT)

#define AS1 __attribute__((address_space(1)))
#define AS3 __attribute__((address_space(3)))

#if defined(__has_builtin)
#if __has_builtin(__builtin_amdgcn_global_load_async_to_lds_b128)
#define USE_ASYNC_LDS 1
#endif
#endif

typedef __attribute__((ext_vector_type(16))) __bf16 v16bf;
typedef __attribute__((ext_vector_type(8)))  float  v8f;
typedef __attribute__((ext_vector_type(4)))  int    v4i;

__device__ __forceinline__ void wait_asynccnt0() {
#if defined(USE_ASYNC_LDS)
#if __has_builtin(__builtin_amdgcn_s_wait_asynccnt)
  __builtin_amdgcn_s_wait_asynccnt(0);
#else
  asm volatile("s_wait_asynccnt 0" ::: "memory");
#endif
#endif
}

#if defined(USE_ASYNC_LDS)
// One b128 async DMA: global fp32 -> LDS. Builtin takes v4i AS1* / v4i AS3*.
__device__ __forceinline__ void async_b128(const float* g, float* l) {
  __builtin_amdgcn_global_load_async_to_lds_b128((AS1 v4i*)g, (AS3 v4i*)l, 0, 0);
}
#endif

// Register-resident sorted (ascending) top-K insertion; all indices constant
// after unrolling so ts/ti stay in VGPRs (no scratch).
__device__ __forceinline__ void topk_insert(float (&ts)[TOPK], int (&ti)[TOPK],
                                            float s, int idx) {
  if (s <= ts[0]) return;          // ts[0] is current minimum (early out)
  bool placed = false;
#pragma unroll
  for (int u = 0; u < TOPK - 1; ++u) {
    if (!placed) {
      if (s > ts[u + 1]) { ts[u] = ts[u + 1]; ti[u] = ti[u + 1]; }
      else               { ts[u] = s;         ti[u] = idx; placed = true; }
    }
  }
  if (!placed) { ts[TOPK - 1] = s; ti[TOPK - 1] = idx; }
}

// ---------------------------------------------------------------------------
// Phase 1: per-block streaming GEMM (bf16 WMMA) + per-query partial top-16.
// Block = 256 threads = 8 waves; wave w owns queries [32w, 32w+32).
// ---------------------------------------------------------------------------
__global__ __launch_bounds__(256) void sim_topk_partial(
    const float* __restrict__ Q,          // [256][128]
    const float* __restrict__ bank,       // [1e6][128]
    const long long* __restrict__ excl,   // [256]
    float* __restrict__ part_score,       // [NBLK][256][16]
    int*   __restrict__ part_idx) {       // [NBLK][256][16]
#if defined(USE_ASYNC_LDS)
  __shared__ float  lds_stage[ROWS_IT][EMB];     // 16 KB fp32 async landing pad
#endif
  __shared__ __bf16 lds_bank[ROWS_IT][EMB];      // 8 KB bf16 WMMA tile
  __shared__ float  lds_sim[NQ][ROWS_IT + 1];    // padded vs LDS bank conflicts

  const int tid   = threadIdx.x;
  const int wave  = tid >> 5;
  const int lane  = tid & 31;
  const int lrow  = lane & 15;
  const int khalf = (lane >> 4) << 3;    // 0 for lanes 0-15, 8 for lanes 16-31

  // Per-thread slice of the staged tile: row sr, 16 consecutive columns.
  const int sr  = tid >> 3;
  const int sc0 = (tid & 7) * 16;

  // Preload this wave's two 16-query A tiles as bf16 WMMA fragments.
  // 16-bit A layout (16x32): lane<16 holds K = {c*32+0..7, c*32+16..23},
  // lane>=16 the +8 shifted ranges.
  v16bf afrag[2][4];
#pragma unroll
  for (int t = 0; t < 2; ++t) {
    const float* qrow = Q + (wave * 32 + t * 16 + lrow) * EMB;
#pragma unroll
    for (int c = 0; c < 4; ++c) {
      v16bf a;
#pragma unroll
      for (int i = 0; i < 16; ++i) {
        int k = c * 32 + khalf + ((i < 8) ? i : (i + 8));
        a[i] = (__bf16)qrow[k];
      }
      afrag[t][c] = a;
    }
  }

  float ts[TOPK]; int ti[TOPK];
#pragma unroll
  for (int u = 0; u < TOPK; ++u) { ts[u] = -__builtin_inff(); ti[u] = -1; }

  const long long ex   = excl[tid];      // thread tid owns query tid
  const long long base = (long long)blockIdx.x * CHUNK;

#if defined(USE_ASYNC_LDS)
  // ---- prologue: async-DMA tile 0 into the fp32 staging buffer ----
  {
    long long grow = base + sr;
    if (grow >= BANK_N) grow = BANK_N - 1;        // clamp (masked in scan)
    const float* src = bank + grow * (long long)EMB + sc0;
    float*       dst = &lds_stage[sr][sc0];
#pragma unroll
    for (int j = 0; j < 4; ++j) async_b128(src + j * 4, dst + j * 4);
  }
#endif

  for (int it = 0; it < NIT; ++it) {
    const long long rowbase = base + (long long)it * ROWS_IT;

#if defined(USE_ASYNC_LDS)
    wait_asynccnt0();                  // fp32 tile `it` landed in lds_stage
    __syncthreads();
    // ---- convert staged fp32 tile -> bf16 WMMA tile ----
#pragma unroll
    for (int j = 0; j < 16; ++j)
      lds_bank[sr][sc0 + j] = (__bf16)lds_stage[sr][sc0 + j];
    __syncthreads();                   // lds_bank ready; lds_stage free
    // ---- kick off DMA for tile it+1, overlapping WMMA + scan below ----
    if (it + 1 < NIT) {
      long long grow = rowbase + ROWS_IT + sr;
      if (grow >= BANK_N) grow = BANK_N - 1;
      const float* src = bank + grow * (long long)EMB + sc0;
      float*       dst = &lds_stage[sr][sc0];
#pragma unroll
      for (int j = 0; j < 4; ++j) async_b128(src + j * 4, dst + j * 4);
    }
#else
    // ---- fallback: synchronous stage global -> regs -> cvt -> LDS ----
    {
      long long grow = rowbase + sr;
      if (grow >= BANK_N) grow = BANK_N - 1;      // clamp (masked in scan)
      const float4* src = (const float4*)(bank + grow * (long long)EMB + sc0);
#pragma unroll
      for (int j = 0; j < 4; ++j) {
        float4 v = src[j];
        lds_bank[sr][sc0 + j * 4 + 0] = (__bf16)v.x;
        lds_bank[sr][sc0 + j * 4 + 1] = (__bf16)v.y;
        lds_bank[sr][sc0 + j * 4 + 2] = (__bf16)v.z;
        lds_bank[sr][sc0 + j * 4 + 3] = (__bf16)v.w;
      }
    }
    __syncthreads();
#endif

    // ---- WMMA: 2 bank n-tiles x 2 query tiles x (K=128/32) ----
#pragma unroll
    for (int nt = 0; nt < 2; ++nt) {
      v16bf bfrag[4];
      const int n = nt * 16 + lrow;      // B column = bank row in tile
#pragma unroll
      for (int c = 0; c < 4; ++c) {
        v16bf bv;
#pragma unroll
        for (int i = 0; i < 16; ++i) {
          int k = c * 32 + khalf + ((i < 8) ? i : (i + 8));
          bv[i] = lds_bank[n][k];
        }
        bfrag[c] = bv;
      }
#pragma unroll
      for (int t = 0; t < 2; ++t) {
        v8f acc = {};
#pragma unroll
        for (int c = 0; c < 4; ++c)
          acc = __builtin_amdgcn_wmma_f32_16x16x32_bf16(
              false, afrag[t][c], false, bfrag[c], (short)0, acc, false, false);
        // C/D layout: VGPR r, lane<16 -> (M=r, N=lane); lane>=16 -> (M=8+r)
        const int mrow = wave * 32 + t * 16 + khalf;
#pragma unroll
        for (int rI = 0; rI < 8; ++rI)
          lds_sim[mrow + rI][nt * 16 + lrow] = acc[rI];
      }
    }
    __syncthreads();

    // ---- thread tid scans the 32 fresh sims of its own query ----
    const bool full_tile = (rowbase + ROWS_IT) <= (long long)BANK_N;
    const bool ex_here   = (ex >= rowbase) && (ex < rowbase + ROWS_IT);
    if (full_tile && !ex_here) {
#pragma unroll 4
      for (int j = 0; j < ROWS_IT; ++j)
        topk_insert(ts, ti, lds_sim[tid][j], (int)(rowbase + j));
    } else {
#pragma unroll 4
      for (int j = 0; j < ROWS_IT; ++j) {
        long long grow = rowbase + j;
        float s = lds_sim[tid][j];
        if (grow < BANK_N && grow != ex) topk_insert(ts, ti, s, (int)grow);
      }
    }
    __syncthreads();
  }

  // emit partials sorted descending
  float* ps = part_score + ((long long)blockIdx.x * NQ + tid) * TOPK;
  int*   pi = part_idx   + ((long long)blockIdx.x * NQ + tid) * TOPK;
#pragma unroll
  for (int u = 0; u < TOPK; ++u) { ps[u] = ts[TOPK - 1 - u]; pi[u] = ti[TOPK - 1 - u]; }
}

// ---------------------------------------------------------------------------
// Phase 2: one wave per query merges NBLK partial top-16 lists, emits sorted
// top-16 indices (+ gathers neighbor embeddings).
// ---------------------------------------------------------------------------
__global__ __launch_bounds__(32) void topk_merge(
    const float* __restrict__ part_score, const int* __restrict__ part_idx,
    const float* __restrict__ bank,
    float* __restrict__ out_emb,          // [256][16][128]
    float* __restrict__ out_idx) {        // [256][16] (indices as f32)
  const int q    = blockIdx.x;
  const int lane = threadIdx.x;

  float ts[TOPK]; int ti[TOPK];
#pragma unroll
  for (int u = 0; u < TOPK; ++u) { ts[u] = -__builtin_inff(); ti[u] = -1; }

  const int NC = NBLK * TOPK;
  for (int j = lane; j < NC; j += 32) {
    int blk = j / TOPK, u = j - blk * TOPK;
    long long o = ((long long)blk * NQ + q) * TOPK + u;
    topk_insert(ts, ti, part_score[o], part_idx[o]);
  }

  for (int r = 0; r < TOPK; ++r) {
    // per-lane max (register scan, constant indices)
    float lm = -__builtin_inff(); int lpos = -1;
#pragma unroll
    for (int u = 0; u < TOPK; ++u)
      if (ts[u] > lm) { lm = ts[u]; lpos = u; }
    // wave-wide argmax via xor-shuffle reduction (wave32)
    float bm = lm; int bl = lane;
#pragma unroll
    for (int off = 16; off > 0; off >>= 1) {
      float om = __shfl_xor(bm, off, 32);
      int   ol = __shfl_xor(bl, off, 32);
      if (om > bm) { bm = om; bl = ol; }
    }
    int widx = 0;
    if (lane == bl) {
#pragma unroll
      for (int u = 0; u < TOPK; ++u)
        if (u == lpos) { widx = ti[u]; ts[u] = -__builtin_inff(); }
    }
    widx = __shfl(widx, bl, 32);
    if (lane == 0) out_idx[q * TOPK + r] = (float)widx;
    // gather neighbor embedding row
    const float* brow = bank + (long long)widx * EMB;
    float* orow = out_emb + ((long long)q * TOPK + r) * EMB;
#pragma unroll
    for (int c = 0; c < EMB / 32; ++c)
      orow[lane + c * 32] = brow[lane + c * 32];
  }
}

extern "C" void kernel_launch(void* const* d_in, const int* in_sizes, int n_in,
                              void* d_out, int out_size, void* d_ws, size_t ws_size,
                              hipStream_t stream) {
  (void)in_sizes; (void)n_in; (void)out_size; (void)ws_size;
  const float*     Qv   = (const float*)d_in[0];
  const float*     bank = (const float*)d_in[1];
  const long long* excl = (const long long*)d_in[2];
  // d_in[3] = k (compile-time TOPK)

  float* part_score = (float*)d_ws;
  int*   part_idx   = (int*)((char*)d_ws + sizeof(float) * (size_t)NBLK * NQ * TOPK);

  sim_topk_partial<<<NBLK, 256, 0, stream>>>(Qv, bank, excl, part_score, part_idx);

  float* out_emb = (float*)d_out;
  float* out_idx = out_emb + (size_t)NQ * TOPK * EMB;
  topk_merge<<<NQ, 32, 0, stream>>>(part_score, part_idx, bank, out_emb, out_idx);
}
